// SelfAttention_20246475833997
// MI455X (gfx1250) — matrix operand order repaired and checked by reading, not compiled, not run on hardware
//
#include <hip/hip_runtime.h>

typedef __attribute__((ext_vector_type(16))) __bf16 v16bf;
typedef __attribute__((ext_vector_type(8)))  __bf16 v8bf;
typedef __attribute__((ext_vector_type(8)))  float  v8f;
typedef __attribute__((ext_vector_type(4)))  unsigned int v4u;
typedef __attribute__((ext_vector_type(8)))  int v8i;
typedef __attribute__((ext_vector_type(4)))  int v4i;

#define D_DIM   1024
#define S_DIM   2048
#define B_DIM   4
#define H_DIM   16
#define DH_DIM  64
#define M_TOT   (B_DIM * S_DIM)   /* 8192 */

// ---------------------------------------------------------------- helpers ---

__device__ __forceinline__ v16bf frag_join(v8bf lo, v8bf hi) {
  return __builtin_shufflevector(lo, hi, 0,1,2,3,4,5,6,7,8,9,10,11,12,13,14,15);
}

__device__ __forceinline__ v8f wmma_bf16(v16bf a, v16bf b, v8f c) {
  // (neg_a, A, neg_b, B, c_mod, C, reuse_a, reuse_b)
  return __builtin_amdgcn_wmma_f32_16x16x32_bf16(false, a, false, b,
                                                 (short)0, c, false, false);
}

// reduce across the 16 lanes of a half-wave (xor masks 1,2,4,8 stay in half)
__device__ __forceinline__ float rowred_max(float v) {
#pragma unroll
  for (int m = 1; m < 16; m <<= 1) v = fmaxf(v, __shfl_xor(v, m, 32));
  return v;
}
__device__ __forceinline__ float rowred_sum(float v) {
#pragma unroll
  for (int m = 1; m < 16; m <<= 1) v += __shfl_xor(v, m, 32);
  return v;
}

// Flat shared-aperture addresses carry the wave-relative LDS byte offset in
// the low 32 bits (ISA 10.2).
__device__ __forceinline__ unsigned lds_byte_off(const void* p) {
  return (unsigned)(uintptr_t)p;
}

// --------------------------------------------------------------- TDM load ---
// Issue a Tensor Data Mover 2D tile load: tile_dim0=32 elems (contiguous K),
// tile_dim1=128 rows, element = bf16 (2B), tensor row stride = 1024 elems.
// Descriptor bitfields per CDNA5 ISA section 8.3/8.4.
__device__ __forceinline__ void tdm_load_tile_128x32(unsigned lds_off,
                                                     const __bf16* tile,
                                                     unsigned tensor_rows) {
  unsigned long long ga = (unsigned long long)(uintptr_t)tile;

  v4u g0;
  g0[0] = 1u;                                        // count=1, user descriptor
  g0[1] = lds_off;                                   // lds_addr [63:32]
  g0[2] = (unsigned)ga;                              // global_addr [95:64]
  g0[3] = (unsigned)((ga >> 32) & 0x1FFFFFFu)        // global_addr [120:96]
        | (2u << 30);                                // type=2 ("image") [127:126]

  const unsigned d0 = (unsigned)D_DIM;               // tensor_dim0 (K extent)
  const unsigned d1 = tensor_rows;                   // tensor_dim1 (row extent)
  v8i g1;
  g1[0] = (int)(1u << 16);                           // wg_mask=0, data_size=1 (2B)
  g1[1] = (int)((d0 & 0xFFFFu) << 16);               // abar=0 | tensor_dim0[15:0]
  g1[2] = (int)((d0 >> 16) | ((d1 & 0xFFFFu) << 16));// dim0[31:16] | dim1[15:0]
  g1[3] = (int)((d1 >> 16) | (32u << 16));           // dim1[31:16] | tile_dim0=32
  g1[4] = (int)128u;                                 // tile_dim1=128, tile_dim2=0
  g1[5] = (int)(unsigned)D_DIM;                      // tensor_dim0_stride[31:0]
  g1[6] = 0;                                         // stride0[47:32] | stride1[15:0]
  g1[7] = 0;                                         // stride1[47:16]
  v4i gz4 = {0, 0, 0, 0};
  v8i gz8 = {0, 0, 0, 0, 0, 0, 0, 0};
  __builtin_amdgcn_tensor_load_to_lds(g0, g1, gz4, gz4, gz8, 0);
}

// ----------------------------------------------------- precision conversion -

__global__ __launch_bounds__(256)
void convert_x_kernel(const float* __restrict__ x, __bf16* __restrict__ xb, int n) {
  int stride = gridDim.x * blockDim.x;
  for (int i = blockIdx.x * blockDim.x + threadIdx.x; i < n; i += stride)
    xb[i] = (__bf16)x[i];
}

// WT[w][n][k] = W_w[k][n]  (store transposed so WMMA B-fragments are contiguous)
__global__ __launch_bounds__(256)
void convert_w_kernel(const float* __restrict__ Wq, const float* __restrict__ Wk,
                      const float* __restrict__ Wv, const float* __restrict__ Wo,
                      __bf16* __restrict__ WT) {
  int i = blockIdx.x * blockDim.x + threadIdx.x;   // over D*D
  int w = blockIdx.y;
  const float* W = (w == 0) ? Wq : (w == 1) ? Wk : (w == 2) ? Wv : Wo;
  int n = i >> 10, k = i & (D_DIM - 1);
  WT[(size_t)w * D_DIM * D_DIM + i] = (__bf16)W[(size_t)k * D_DIM + n];
}

// ------------------------------------------------------------- GEMM core ----
// 128x128 block tile, 256 threads (8 waves), wave = 32(M) x 64(N), K-step 32.
// TDM DMA fills the next LDS buffer while all waves run WMMAs on the current.

__device__ __forceinline__ void gemm_tile_128x128(
    const __bf16* __restrict__ Xb, const __bf16* __restrict__ WT,
    int m0, int n0, unsigned xrows, v8f acc[2][4],
    __bf16 (*lx)[128 * 32], __bf16 (*lw)[128 * 32]) {
  const int tid  = threadIdx.x;
  const int lane = tid & 31;
  const int wid  = tid >> 5;
  const int wm = wid >> 1, wn = wid & 1;
  const int mrow = lane & 15;
  const int half = lane >> 4;

  const v8f zero = {0.f, 0.f, 0.f, 0.f, 0.f, 0.f, 0.f, 0.f};
#pragma unroll
  for (int i = 0; i < 2; i++)
#pragma unroll
    for (int j = 0; j < 4; j++) acc[i][j] = zero;

  const __bf16* xtile = Xb + (size_t)m0 * D_DIM;
  const __bf16* wtile = WT + (size_t)n0 * D_DIM;

  // prologue: DMA first K-slab into buffer 0
  if (wid == 0) {
    tdm_load_tile_128x32(lds_byte_off(&lx[0][0]), xtile, xrows);
    tdm_load_tile_128x32(lds_byte_off(&lw[0][0]), wtile, (unsigned)D_DIM);
    __builtin_amdgcn_s_wait_tensorcnt(0);
  }
  __syncthreads();

  int buf = 0;
  for (int k0 = 0; k0 < D_DIM; k0 += 32) {
    // prefetch next K-slab into the other buffer (overlaps with WMMAs below)
    if ((k0 + 32 < D_DIM) && wid == 0) {
      tdm_load_tile_128x32(lds_byte_off(&lx[buf ^ 1][0]), xtile + k0 + 32, xrows);
      tdm_load_tile_128x32(lds_byte_off(&lw[buf ^ 1][0]), wtile + k0 + 32,
                           (unsigned)D_DIM);
    }

    const __bf16* cx = lx[buf];
    const __bf16* cw = lw[buf];
    v16bf a[2], bm[4];
#pragma unroll
    for (int i = 0; i < 2; i++) {           // A: rows m, K runs {8h.., 16+8h..}
      int r = wm * 32 + i * 16 + mrow;
      a[i] = frag_join(*(const v8bf*)&cx[r * 32 + 8 * half],
                       *(const v8bf*)&cx[r * 32 + 16 + 8 * half]);
    }
#pragma unroll
    for (int j = 0; j < 4; j++) {           // B: col n, K runs {8h.., 16+8h..}
      int r = wn * 64 + j * 16 + mrow;
      bm[j] = frag_join(*(const v8bf*)&cw[r * 32 + 8 * half],
                        *(const v8bf*)&cw[r * 32 + 16 + 8 * half]);
    }
#pragma unroll
    for (int i = 0; i < 2; i++)
#pragma unroll
      for (int j = 0; j < 4; j++)
        acc[i][j] = wmma_bf16(a[i], bm[j], acc[i][j]);

    if (wid == 0) __builtin_amdgcn_s_wait_tensorcnt(0);
    __syncthreads();
    buf ^= 1;
  }
}

// --------------------------------------------------------- QKV projection --

__global__ __launch_bounds__(256)
void qkv_gemm_kernel(const __bf16* __restrict__ xb, const __bf16* __restrict__ WT3,
                     const float* __restrict__ bq, const float* __restrict__ bk,
                     const float* __restrict__ bv,
                     __bf16* __restrict__ Qh, __bf16* __restrict__ Kh,
                     __bf16* __restrict__ Vt) {
  __shared__ __align__(16) __bf16 lx[2][128 * 32];
  __shared__ __align__(16) __bf16 lw[2][128 * 32];
  const int m0 = blockIdx.x * 128;
  const int n0 = blockIdx.y * 128;
  const int z  = blockIdx.z;                         // 0=Q 1=K 2=V
  const __bf16* WT = WT3 + (size_t)z * D_DIM * D_DIM;
  const float* bias = (z == 0) ? bq : (z == 1) ? bk : bv;

  v8f acc[2][4];
  gemm_tile_128x128(xb, WT, m0, n0, (unsigned)M_TOT, acc, lx, lw);

  const int tid  = threadIdx.x;
  const int lane = tid & 31;
  const int wid  = tid >> 5;
  const int wm = wid >> 1, wn = wid & 1;
  const int nlane = lane & 15, half = lane >> 4;
  const float qscale = 0.125f;                        // 1/sqrt(DH)

#pragma unroll
  for (int i = 0; i < 2; i++) {
#pragma unroll
    for (int j = 0; j < 4; j++) {
      int ng = n0 + wn * 64 + j * 16 + nlane;
      int h = ng >> 6, dh = ng & 63;
      float bb = bias[ng];
#pragma unroll
      for (int r = 0; r < 8; r++) {
        int mg = m0 + wm * 32 + i * 16 + r + 8 * half;
        int bidx = mg >> 11, s = mg & (S_DIM - 1);
        size_t bh = (size_t)(bidx * H_DIM + h);
        float v = acc[i][j][r] + bb;
        if (z == 0)
          Qh[(bh * S_DIM + s) * DH_DIM + dh] = (__bf16)(v * qscale);
        else if (z == 1)
          Kh[(bh * S_DIM + s) * DH_DIM + dh] = (__bf16)v;
        else
          Vt[(bh * DH_DIM + dh) * S_DIM + s] = (__bf16)v;   // V stored transposed
      }
    }
  }
}

// ---------------------------------------------------- flash attention core --
// One wave per 16-query tile; stream keys in chunks of 32; online softmax.

__global__ __launch_bounds__(128)
void attn_kernel(const __bf16* __restrict__ Q, const __bf16* __restrict__ K,
                 const __bf16* __restrict__ Vt, __bf16* __restrict__ ctxb) {
  __shared__ __align__(16) __bf16 pbuf[4][16 * 32];
  const int tid  = threadIdx.x;
  const int lane = tid & 31;
  const int wid  = tid >> 5;
  const int gw = blockIdx.x * 4 + wid;               // 0..8191
  const int bh = gw >> 7;                            // (b,h) pair
  const int q0 = (gw & 127) * 16;
  const int b = bh >> 4, h = bh & 15;
  const int n = lane & 15, half = lane >> 4;

  const __bf16* Qp = Q + ((size_t)bh * S_DIM + q0) * DH_DIM;
  const __bf16* Kp = K + (size_t)bh * S_DIM * DH_DIM;
  const __bf16* Vp = Vt + (size_t)bh * DH_DIM * S_DIM;
  __bf16* pb = pbuf[wid];

  // resident Q A-fragments (softmax scale folded in at projection time)
  v16bf qa[2];
#pragma unroll
  for (int f = 0; f < 2; f++)
    qa[f] = frag_join(*(const v8bf*)&Qp[n * DH_DIM + 32 * f + 8 * half],
                      *(const v8bf*)&Qp[n * DH_DIM + 32 * f + 16 + 8 * half]);

  const v8f zero = {0.f, 0.f, 0.f, 0.f, 0.f, 0.f, 0.f, 0.f};
  v8f o[4];
#pragma unroll
  for (int j = 0; j < 4; j++) o[j] = zero;
  float mrow[8], lrow[8];
#pragma unroll
  for (int r = 0; r < 8; r++) { mrow[r] = -3.0e38f; lrow[r] = 0.f; }

  for (int k0 = 0; k0 < S_DIM; k0 += 32) {
    // logits: two 16x16 tiles (keys k0..+15, k0+16..+31), K-dim = DH = 2x32
    v8f s[2];
#pragma unroll
    for (int t = 0; t < 2; t++) {
      const __bf16* krow = &Kp[(size_t)(k0 + 16 * t + n) * DH_DIM];
      v16bf b0 = frag_join(*(const v8bf*)&krow[8 * half],
                           *(const v8bf*)&krow[16 + 8 * half]);
      v16bf b1 = frag_join(*(const v8bf*)&krow[32 + 8 * half],
                           *(const v8bf*)&krow[32 + 16 + 8 * half]);
      v8f z = wmma_bf16(qa[0], b0, zero);
      s[t] = wmma_bf16(qa[1], b1, z);
    }

    // online softmax (row = r + 8*half, striped over 16 lanes of the half)
    float alpha[8];
#pragma unroll
    for (int r = 0; r < 8; r++) {
      float mx = rowred_max(fmaxf(s[0][r], s[1][r]));
      float mnew = fmaxf(mrow[r], mx);
      alpha[r] = __expf(mrow[r] - mnew);
      float p0 = __expf(s[0][r] - mnew);
      float p1 = __expf(s[1][r] - mnew);
      s[0][r] = p0; s[1][r] = p1;
      lrow[r] = lrow[r] * alpha[r] + rowred_sum(p0 + p1);
      mrow[r] = mnew;
    }

    // repack probs: C-layout -> LDS -> A-layout fragment
#pragma unroll
    for (int t = 0; t < 2; t++)
#pragma unroll
      for (int r = 0; r < 8; r++)
        pb[(r + 8 * half) * 32 + 16 * t + n] = (__bf16)s[t][r];
    asm volatile("s_wait_dscnt 0" ::: "memory");
    v16bf pa = frag_join(*(const v8bf*)&pb[n * 32 + 8 * half],
                         *(const v8bf*)&pb[n * 32 + 16 + 8 * half]);

    // rescale O and accumulate P*V (V^T gives contiguous B fragments)
#pragma unroll
    for (int j = 0; j < 4; j++) {
#pragma unroll
      for (int r = 0; r < 8; r++) o[j][r] *= alpha[r];
      const __bf16* vrow = &Vp[(size_t)(j * 16 + n) * S_DIM + k0];
      v16bf vb = frag_join(*(const v8bf*)&vrow[8 * half],
                           *(const v8bf*)&vrow[16 + 8 * half]);
      o[j] = wmma_bf16(pa, vb, o[j]);
    }
  }

  // normalize + scatter to [B*S, D] bf16 for the output projection
#pragma unroll
  for (int j = 0; j < 4; j++) {
#pragma unroll
    for (int r = 0; r < 8; r++) {
      size_t row = (size_t)b * S_DIM + q0 + r + 8 * half;
      int col = h * DH_DIM + j * 16 + n;
      ctxb[row * D_DIM + col] = (__bf16)(o[j][r] / lrow[r]);
    }
  }
}

// ------------------------------------------------- output proj + residual --

__global__ __launch_bounds__(256)
void oproj_gemm_kernel(const __bf16* __restrict__ ctxb, const __bf16* __restrict__ WTo,
                       const float* __restrict__ bo, const float* __restrict__ x,
                       float* __restrict__ out) {
  __shared__ __align__(16) __bf16 lx[2][128 * 32];
  __shared__ __align__(16) __bf16 lw[2][128 * 32];
  const int m0 = blockIdx.x * 128;
  const int n0 = blockIdx.y * 128;

  v8f acc[2][4];
  gemm_tile_128x128(ctxb, WTo, m0, n0, (unsigned)M_TOT, acc, lx, lw);

  const int tid  = threadIdx.x;
  const int lane = tid & 31;
  const int wid  = tid >> 5;
  const int wm = wid >> 1, wn = wid & 1;
  const int nlane = lane & 15, half = lane >> 4;

#pragma unroll
  for (int i = 0; i < 2; i++) {
#pragma unroll
    for (int j = 0; j < 4; j++) {
      int ng = n0 + wn * 64 + j * 16 + nlane;
      float bb = bo[ng];
#pragma unroll
      for (int r = 0; r < 8; r++) {
        size_t mg = (size_t)(m0 + wm * 32 + i * 16 + r + 8 * half);
        out[mg * D_DIM + ng] = acc[i][j][r] + bb + x[mg * D_DIM + ng];
      }
    }
  }
}

// ------------------------------------------------------------------ launch --

extern "C" void kernel_launch(void* const* d_in, const int* in_sizes, int n_in,
                              void* d_out, int out_size, void* d_ws, size_t ws_size,
                              hipStream_t stream) {
  (void)in_sizes; (void)n_in; (void)out_size; (void)ws_size;
  const float* x  = (const float*)d_in[0];
  const float* Wq = (const float*)d_in[1];
  const float* bq = (const float*)d_in[2];
  const float* Wk = (const float*)d_in[3];
  const float* bk = (const float*)d_in[4];
  const float* Wv = (const float*)d_in[5];
  const float* bv = (const float*)d_in[6];
  const float* Wo = (const float*)d_in[7];
  const float* bo = (const float*)d_in[8];
  float* out = (float*)d_out;

  char* ws = (char*)d_ws;
  __bf16* xb  = (__bf16*)(ws);                         // 16 MB: [8192][1024]
  __bf16* WT  = (__bf16*)(ws + ((size_t)16 << 20));    //  8 MB: 4x[1024][1024] (transposed)
  __bf16* Qh  = (__bf16*)(ws + ((size_t)24 << 20));    // 16 MB: [B,H,S,DH]
  __bf16* Kh  = (__bf16*)(ws + ((size_t)40 << 20));    // 16 MB: [B,H,S,DH]
  __bf16* Vt  = (__bf16*)(ws + ((size_t)56 << 20));    // 16 MB: [B,H,DH,S]
  __bf16* ctx = (__bf16*)(ws + ((size_t)72 << 20));    // 16 MB: [8192][1024]

  convert_x_kernel<<<dim3(4096), dim3(256), 0, stream>>>(x, xb, M_TOT * D_DIM);
  convert_w_kernel<<<dim3(4096, 4), dim3(256), 0, stream>>>(Wq, Wk, Wv, Wo, WT);
  qkv_gemm_kernel<<<dim3(64, 8, 3), dim3(256), 0, stream>>>(xb, WT, bq, bk, bv,
                                                            Qh, Kh, Vt);
  attn_kernel<<<dim3(2048), dim3(128), 0, stream>>>(Qh, Kh, Vt, ctx);
  oproj_gemm_kernel<<<dim3(64, 8), dim3(256), 0, stream>>>(
      ctx, WT + (size_t)3 * D_DIM * D_DIM, bo, x, out);
}
